// SAGE_23862838297295
// MI455X (gfx1250) — matrix operand order repaired and checked by reading, not compiled, not run on hardware
//
#include <hip/hip_runtime.h>
#include <hip/hip_bf16.h>

#define N_NODES  32768
#define N_GRAPHS 256
#define DEG      16
#define IN_F     128
#define HID_F    256
#define OUT_F    16

typedef __bf16 bf16;
typedef __attribute__((ext_vector_type(16))) __bf16 v16bf;
typedef __attribute__((ext_vector_type(8)))  __bf16 v8bf;
typedef __attribute__((ext_vector_type(8)))  float  v8f;

// Build a 16-element bf16 A/B fragment from two contiguous 16-byte chunks.
static __device__ __forceinline__ v16bf load_frag32(const bf16* p0, const bf16* p1) {
  v8bf lo = *(const v8bf*)p0;
  v8bf hi = *(const v8bf*)p1;
  return __builtin_shufflevector(lo, hi, 0,1,2,3,4,5,6,7,8,9,10,11,12,13,14,15);
}

// Branch-free activations on v_exp_f32 / v_rcp_f32 (no libm polynomial paths).
static __device__ __forceinline__ float sigm(float x) {
  return __builtin_amdgcn_rcpf(1.0f + __expf(-x));
}
static __device__ __forceinline__ float fast_tanh(float x) {
  // 1 - 2/(e^{2x}+1): exp overflow -> +1, underflow -> -1 (correct saturation).
  return 1.0f - 2.0f * __builtin_amdgcn_rcpf(__expf(2.0f * x) + 1.0f);
}

// Accumulate: acc[0..NT) += Arow(16 x F) * W-slice(F x 16*NT).
// A layout (16-bit A 16x32, ISA 7.12.2): lane<16 row=lane holds K {k0..k0+7, k0+16..k0+23},
// lane>=16 row=lane-16 holds K {k0+8..k0+15, k0+24..k0+31} -> two contiguous 16B loads.
// B layout (16-bit B 32x16): lane<16 col=lane holds K {k0..k0+7, k0+16..k0+23}, lane>=16
// col=lane-16 holds K {k0+8..k0+15, k0+24..k0+31} -> two 16B loads from W row-major [G, F].
// Software-pipelined one fragment deep: the load for step i+1 is issued before the WMMA
// of step i, so the backend can wait on loadcnt<=1 instead of a full drain per WMMA.
// aRow may point to global (gather) or LDS (recurrent h) -- inlining resolves the space.
template<int F, int NT>
static __device__ __forceinline__ void mma_part(
    const bf16* __restrict__ aRow,   // this lane's source row
    const bf16* __restrict__ W,      // [*, F] row-major bf16
    int col0,                        // column of tile 0 for this lane
    int hi,                          // lane>>4
    v8f* acc)
{
  constexpr int KC  = F / 32;
  constexpr int TOT = KC * NT;
  auto loadA = [&](int kc) {
    const bf16* p = aRow + kc * 32 + hi * 8;
    return load_frag32(p, p + 16);
  };
  auto loadB = [&](int kc, int n) {
    const bf16* bp = W + (size_t)(col0 + n * 16) * F + (kc * 32 + hi * 8);
    return load_frag32(bp, bp + 16);
  };
  v16bf a_cur = loadA(0);
  v16bf b_cur = loadB(0, 0);
#pragma unroll
  for (int i = 0; i < TOT; ++i) {
    const int n = i % NT;
    v16bf a_nxt = a_cur;
    v16bf b_nxt = b_cur;
    if (i + 1 < TOT) {
      const int kcn = (i + 1) / NT;
      const int nn  = (i + 1) % NT;
      if (nn == 0) a_nxt = loadA(kcn);
      b_nxt = loadB(kcn, nn);
    }
    acc[n] = __builtin_amdgcn_wmma_f32_16x16x32_bf16(
        false, a_cur, false, b_cur, (short)0, acc[n], false, false);
    a_cur = a_nxt;
    b_cur = b_nxt;
  }
}

// Fully fused SAGEConv('lstm') layer: 16 LSTM steps + fc_self/fc_neigh + bias + ReLU.
// One WG owns 64 nodes end-to-end: c stays in VGPRs, h stays in LDS (bf16, padded rows),
// gates go through LDS once per step so i/f/g/o (held by different waves) combine.
template<int F>
__global__ __launch_bounds__(512, 1)
void sage_conv_kernel(const bf16*  __restrict__ featB,  // [N, F] layer input (constant)
                      const int*   __restrict__ nbrs,   // [N, DEG]
                      const bf16*  __restrict__ WiB,    // [4F, F] bf16 (natural layout)
                      const bf16*  __restrict__ WhB,    // [4F, F]
                      const float* __restrict__ bi,     // [4F]
                      const float* __restrict__ bh,     // [4F]
                      const bf16*  __restrict__ WsB,    // [256, F]
                      const bf16*  __restrict__ WnB,    // [256, F]
                      const float* __restrict__ bfc,    // [256]
                      float* __restrict__ outF,         // [N, 256]
                      bf16*  __restrict__ outB)         // [N, 256]
{
  constexpr int G    = 4 * F;
  constexpr int NT   = G / 64;       // gate tiles per wave
  constexpr int HSTR = F + 8;        // padded h row: +16B so A-frag ds_load_b128 is bank-clean
  constexpr int CPT  = 64 * F / 512; // cell-state elems per thread (step-invariant partition)
  constexpr int LOGF = (F == 128) ? 7 : 8;

  extern __shared__ char smem[];
  float* gates = (float*)smem;                                    // 64*G f32
  bf16*  hLds  = (bf16*)(smem + (size_t)64 * G * 4);              // 64*HSTR bf16
  float* bsum  = (float*)((char*)hLds + (size_t)64 * HSTR * 2);   // G f32 (bi+bh)
  int*   s_nbr = (int*)((char*)bsum + (size_t)G * 4);             // 64*DEG ints

  const int node0 = blockIdx.x * 64;

  // Stage neighbor block (contiguous), combined bias, zero the h tile.
  for (int i = threadIdx.x; i < 64 * DEG; i += 512)
    s_nbr[i] = nbrs[(size_t)node0 * DEG + i];
  for (int i = threadIdx.x; i < G; i += 512)
    bsum[i] = bi[i] + bh[i];
  for (int i = threadIdx.x; i < 64 * HSTR; i += 512)
    hLds[i] = (bf16)0.0f;

  float cReg[CPT];
#pragma unroll
  for (int k = 0; k < CPT; ++k) cReg[k] = 0.0f;
  __syncthreads();

  const int lane = threadIdx.x & 31;
  const int wave = threadIdx.x >> 5;
  const int r    = wave & 3;        // row tile (16 rows each)
  const int cg   = wave >> 2;       // column group: exactly one gate (i/f/g/o)
  const int hi   = lane >> 4;
  const int ml   = r * 16 + (lane & 15);        // local row this lane feeds A with
  const int col0 = cg * (G / 4) + (lane & 15);  // lane column of gate tile 0
  const bf16* hRow = hLds + (size_t)ml * HSTR;

#pragma unroll 1
  for (int t = 0; t < DEG; ++t) {
    // Opaque memory clobber: stops LICM from hoisting the loop-invariant weight
    // fragments out of the t-loop into registers (which spilled WMMA operands to
    // scratch). Weights re-stream from L2 each step and pipeline against the WMMAs.
    asm volatile("" ::: "memory");

    v8f acc[NT];
#pragma unroll
    for (int n = 0; n < NT; ++n)
#pragma unroll
      for (int v = 0; v < 8; ++v) acc[n][v] = 0.0f;

    const bf16* aX = featB + (size_t)s_nbr[ml * DEG + t] * F;  // gathered neighbor row
    mma_part<F, NT>(aX, WiB, col0, hi, acc);                   // global A, global B
    mma_part<F, NT>(hRow, WhB, col0, hi, acc);                 // LDS A, global B

    // Accumulators -> LDS gates. C/D layout: VGPR v -> row hi*8+v, col = lane&15.
    const int rbase = r * 16 + hi * 8;
#pragma unroll
    for (int n = 0; n < NT; ++n) {
      const int col = col0 + n * 16;
#pragma unroll
      for (int v = 0; v < 8; ++v)
        gates[(size_t)(rbase + v) * G + col] = acc[n][v];
    }
    __syncthreads();

    // Fused LSTM cell update (gate order i,f,g,o); c in VGPRs, new h -> LDS.
#pragma unroll
    for (int k = 0; k < CPT; ++k) {
      const int e = threadIdx.x + k * 512;
      const int m = e >> LOGF;
      const int j = e & (F - 1);
      const float gi = gates[(size_t)m * G + j]         + bsum[j];
      const float gf = gates[(size_t)m * G + F + j]     + bsum[F + j];
      const float gg = gates[(size_t)m * G + 2 * F + j] + bsum[2 * F + j];
      const float go = gates[(size_t)m * G + 3 * F + j] + bsum[3 * F + j];
      const float c = sigm(gf) * cReg[k] + sigm(gi) * fast_tanh(gg);
      cReg[k] = c;
      hLds[(size_t)m * HSTR + j] = (bf16)(sigm(go) * fast_tanh(c));
    }
    __syncthreads();
  }

  // SAGE combine epilogue: out = relu(feat*Ws^T + hT*Wn^T + b). G2=256, 4 tiles/wave.
  constexpr int NT2 = 4;
  const int col0b = cg * 64 + (lane & 15);
  v8f acc2[NT2];
#pragma unroll
  for (int n = 0; n < NT2; ++n)
#pragma unroll
    for (int v = 0; v < 8; ++v) acc2[n][v] = 0.0f;

  mma_part<F, NT2>(featB + (size_t)(node0 + ml) * F, WsB, col0b, hi, acc2);
  mma_part<F, NT2>(hRow, WnB, col0b, hi, acc2);

  const int rbase = node0 + r * 16 + hi * 8;
#pragma unroll
  for (int n = 0; n < NT2; ++n) {
    const int col = col0b + n * 16;
    const float bias = bfc[col];
#pragma unroll
    for (int v = 0; v < 8; ++v) {
      float xv = fmaxf(acc2[n][v] + bias, 0.0f);
      const size_t idx = (size_t)(rbase + v) * HID_F + col;
      outF[idx] = xv;
      outB[idx] = (bf16)xv;
    }
  }
}

__global__ void f32_to_bf16_kernel(const float* __restrict__ src, bf16* __restrict__ dst, int n) {
  const int i = blockIdx.x * 256 + threadIdx.x;
  if (i < n) dst[i] = (bf16)src[i];
}

// Per-graph mean pool over 128 nodes + 256->16 classifier.
__global__ __launch_bounds__(256)
void pool_fc_kernel(const float* __restrict__ h2,    // [N, 256]
                    const float* __restrict__ Wc,    // [16, 256]
                    const float* __restrict__ bc,    // [16]
                    float* __restrict__ out)         // [256, 16]
{
  __shared__ float sm[HID_F];
  const int g = blockIdx.x;
  const int j = threadIdx.x;
  float s = 0.0f;
  for (int n = 0; n < N_NODES / N_GRAPHS; ++n)
    s += h2[(size_t)(g * (N_NODES / N_GRAPHS) + n) * HID_F + j];
  sm[j] = s * (1.0f / (float)(N_NODES / N_GRAPHS));
  __syncthreads();
  if (j < OUT_F) {
    float a = bc[j];
    for (int k = 0; k < HID_F; ++k) a += Wc[j * HID_F + k] * sm[k];
    out[(size_t)g * OUT_F + j] = a;
  }
}

extern "C" void kernel_launch(void* const* d_in, const int* in_sizes, int n_in,
                              void* d_out, int out_size, void* d_ws, size_t ws_size,
                              hipStream_t stream) {
  (void)in_sizes; (void)n_in; (void)out_size; (void)ws_size;
  const float* x   = (const float*)d_in[0];
  const int*   nb  = (const int*)  d_in[1];
  const float* Wi1 = (const float*)d_in[2];
  const float* Wh1 = (const float*)d_in[3];
  const float* bi1 = (const float*)d_in[4];
  const float* bh1 = (const float*)d_in[5];
  const float* Ws1 = (const float*)d_in[6];
  const float* Wn1 = (const float*)d_in[7];
  const float* b1  = (const float*)d_in[8];
  const float* Wi2 = (const float*)d_in[9];
  const float* Wh2 = (const float*)d_in[10];
  const float* bi2 = (const float*)d_in[11];
  const float* bh2 = (const float*)d_in[12];
  const float* Ws2 = (const float*)d_in[13];
  const float* Wn2 = (const float*)d_in[14];
  const float* b2  = (const float*)d_in[15];
  const float* Wc  = (const float*)d_in[16];
  const float* bc  = (const float*)d_in[17];
  float* out = (float*)d_out;

  char* ws = (char*)d_ws;
  size_t off = 0;
  auto take = [&](size_t bytes) -> char* {
    char* p = ws + off;
    off = (off + bytes + 255) & ~(size_t)255;
    return p;
  };
  bf16*  xb   = (bf16*) take((size_t)N_NODES * IN_F  * 2);   // x in bf16
  float* h1f  = (float*)take((size_t)N_NODES * HID_F * 4);   // conv1 out f32
  bf16*  h1b  = (bf16*) take((size_t)N_NODES * HID_F * 2);   // conv1 out bf16
  float* h2f  = (float*)take((size_t)N_NODES * HID_F * 4);   // conv2 out f32
  bf16*  h2b  = (bf16*) take((size_t)N_NODES * HID_F * 2);   // conv2 out bf16 (sink)
  bf16*  wi1b = (bf16*) take((size_t)4 * IN_F  * IN_F  * 2);
  bf16*  wh1b = (bf16*) take((size_t)4 * IN_F  * IN_F  * 2);
  bf16*  ws1b = (bf16*) take((size_t)HID_F * IN_F  * 2);
  bf16*  wn1b = (bf16*) take((size_t)HID_F * IN_F  * 2);
  bf16*  wi2b = (bf16*) take((size_t)4 * HID_F * HID_F * 2);
  bf16*  wh2b = (bf16*) take((size_t)4 * HID_F * HID_F * 2);
  bf16*  ws2b = (bf16*) take((size_t)HID_F * HID_F * 2);
  bf16*  wn2b = (bf16*) take((size_t)HID_F * HID_F * 2);

  auto cvt = [&](const float* s, bf16* d, int n) {
    f32_to_bf16_kernel<<<(n + 255) / 256, 256, 0, stream>>>(s, d, n);
  };
  cvt(x,   xb,   N_NODES * IN_F);
  cvt(Wi1, wi1b, 4 * IN_F * IN_F);
  cvt(Wh1, wh1b, 4 * IN_F * IN_F);
  cvt(Ws1, ws1b, HID_F * IN_F);
  cvt(Wn1, wn1b, HID_F * IN_F);
  cvt(Wi2, wi2b, 4 * HID_F * HID_F);
  cvt(Wh2, wh2b, 4 * HID_F * HID_F);
  cvt(Ws2, ws2b, HID_F * HID_F);
  cvt(Wn2, wn2b, HID_F * HID_F);

  // LDS: gates(64*4F f32) + h tile(64*(F+8) bf16) + bias(4F f32) + nbr block(64*16 i32)
  const int smem1 = 64 * (4 * IN_F)  * 4 + 64 * (IN_F  + 8) * 2 + (4 * IN_F)  * 4 + 64 * DEG * 4;
  const int smem2 = 64 * (4 * HID_F) * 4 + 64 * (HID_F + 8) * 2 + (4 * HID_F) * 4 + 64 * DEG * 4;
  (void)hipFuncSetAttribute(reinterpret_cast<const void*>(&sage_conv_kernel<IN_F>),
                            hipFuncAttributeMaxDynamicSharedMemorySize, smem1);
  (void)hipFuncSetAttribute(reinterpret_cast<const void*>(&sage_conv_kernel<HID_F>),
                            hipFuncAttributeMaxDynamicSharedMemorySize, smem2);

  const dim3 gGemm(N_NODES / 64), bGemm(512);

  // conv1 (fused 16 LSTM steps + combine + ReLU), then conv2, then pool+classify.
  sage_conv_kernel<IN_F><<<gGemm, bGemm, smem1, stream>>>(
      xb, nb, wi1b, wh1b, bi1, bh1, ws1b, wn1b, b1, h1f, h1b);
  sage_conv_kernel<HID_F><<<gGemm, bGemm, smem2, stream>>>(
      h1b, nb, wi2b, wh2b, bi2, bh2, ws2b, wn2b, b2, h2f, h2b);
  pool_fc_kernel<<<N_GRAPHS, 256, 0, stream>>>(h2f, Wc, bc, out);
}
